// EMAVectorQuantizer_35184372088912
// MI455X (gfx1250) — matrix operand order repaired and checked by reading, not compiled, not run on hardware
//
#include <hip/hip_runtime.h>

#define D 256
#define VQ_BETA 0.25f
#define SROW 272        // z-tile row stride in bf16: 544B = 34*16B (b128-aligned)
#define ZTILE_E (128 * SROW)          // z tile size in ushort elements (69632 B)
#define BROW 264        // B-buffer row stride in bf16: 528B (TDM pad: 512B + 16B)
#define BBUF_E (16 * BROW)            // one B buffer: 8448 B
#define LDS_TOTAL_B ((ZTILE_E + 2 * BBUF_E) * 2)   // 86528 B dynamic LDS

typedef __attribute__((ext_vector_type(16))) __bf16 v16bf;
typedef __attribute__((ext_vector_type(8)))  float  v8f;
typedef __attribute__((ext_vector_type(4)))  unsigned v4u;
typedef __attribute__((ext_vector_type(8)))  int      v8i;
typedef __attribute__((ext_vector_type(4)))  int      v4i;

struct alignas(16) Q4 { unsigned x, y, z, w; };
union Frag { v16bf v; Q4 q[2]; };

__device__ __forceinline__ unsigned short f2bf(float f) {
  unsigned u = __float_as_uint(f);
  u += 0x7FFFu + ((u >> 16) & 1u);   // round-to-nearest-even
  return (unsigned short)(u >> 16);
}

// Issue one TDM transfer: 16 codebook rows (16*256 bf16 = 8KB contiguous in
// global) -> LDS at ldsoff, with +16B padding every 512B so rows land at a
// 528B (conflict-free, b128-aligned) stride.  D# per CDNA5 ISA section 8.
__device__ __forceinline__ void tdm_load_chunk(unsigned ldsoff,
                                               const unsigned short* gsrc) {
  const unsigned long long ga = (unsigned long long)(size_t)gsrc;
  v4u g0;
  g0.x = 1u;                                   // count=1, no gather
  g0.y = ldsoff;                               // lds_addr [63:32]
  g0.z = (unsigned)ga;                         // global_addr[31:0]
  g0.w = (unsigned)((ga >> 32) & 0x01FFFFFFull) | (2u << 30);  // addr[56:32]|type=2
  v8i g1;
  g1[0] = (int)((1u << 16)     // data_size = 2B
              | (1u << 20)     // pad_enable
              | (6u << 22)     // pad_interval: every 1<<6 qwords = 512B
              | (3u << 25));   // pad_amount: 4 DWORDs = 16B
  g1[1] = (int)(4096u << 16);  // tensor_dim0[15:0]=4096 (bits 63:48)
  g1[2] = (int)(1u << 16);     // tensor_dim0[31:16]=0 | tensor_dim1=1
  g1[3] = (int)(4096u << 16);  // tensor_dim1 hi=0 | tile_dim0=4096 elements
  g1[4] = 0;                   // tile_dim1=0 (1-D tile), tile_dim2=0
  g1[5] = 4096;                // tensor_dim0_stride
  g1[6] = 0;
  g1[7] = 0;
  const v4i gz = {0, 0, 0, 0};
#if defined(__clang_major__) && (__clang_major__ >= 23)
  const v8i gz8 = {0, 0, 0, 0, 0, 0, 0, 0};
  __builtin_amdgcn_tensor_load_to_lds(g0, g1, gz, gz, gz8, 0);
#else
  __builtin_amdgcn_tensor_load_to_lds(g0, g1, gz, gz, 0);
#endif
}

// Deterministic block sum (result valid in thread 0). blockDim.x <= 256.
__device__ __forceinline__ float blockSum(float v, float* sm) {
#pragma unroll
  for (int off = 16; off > 0; off >>= 1) v += __shfl_down(v, off, 32);
  const int lane = threadIdx.x & 31, wid = threadIdx.x >> 5;
  const int nw = (blockDim.x + 31) >> 5;
  if (lane == 0) sm[wid] = v;
  __syncthreads();
  v = (wid == 0 && lane < nw) ? sm[lane] : 0.0f;
  if (wid == 0) {
#pragma unroll
    for (int off = 4; off > 0; off >>= 1) v += __shfl_down(v, off, 32);
  }
  return v;
}

// ---------------- Kernel A: weight f32 -> bf16 + ||e||^2 ----------------
__global__ void vq_prep_weights(const float* __restrict__ w,
                                unsigned short* __restrict__ wbf,
                                float* __restrict__ wnorm2) {
  __shared__ float sm[8];
  const int row = blockIdx.x, t = threadIdx.x;
  const float v = w[(size_t)row * D + t];
  wbf[(size_t)row * D + t] = f2bf(v);
  const float s = blockSum(v * v, sm);
  if (t == 0) wnorm2[row] = s;
}

// ---------------- Kernel B: bf16 WMMA distance GEMM + argmin ------------
// 128 threads (4 waves). Block stages 128 z-rows (bf16) in LDS; each wave
// owns TWO 16-row A tiles resident in VGPRs.  The 16-code B chunk is
// DMA'd into double-buffered LDS by the Tensor Data Mover (wave 0 issues,
// TENSORcnt + workgroup barrier publishes), so B fragments are short-
// latency ds_load_b128 and each feeds two WMMAs (two acc chains).
__global__ __launch_bounds__(128) void vq_argmin(
    const float* __restrict__ z, const unsigned short* __restrict__ wbf,
    const float* __restrict__ wnorm2, int* __restrict__ idxout, int Kcodes) {
  extern __shared__ unsigned short zt[];   // [128][SROW] + 2 B buffers
  const int t = threadIdx.x;
  const int row0 = blockIdx.x * 128;
  const int wave = t >> 5, lane = t & 31;
  const int half = lane >> 4, nl = lane & 15;
  const int nchunks = Kcodes >> 4;

  unsigned short* bb0 = zt + ZTILE_E;
  unsigned short* bb1 = zt + ZTILE_E + BBUF_E;
  const unsigned bboff0 = (unsigned)(size_t)(const void*)bb0;
  const unsigned bboff1 = (unsigned)(size_t)(const void*)bb1;

  // Kick off the first two B-chunk DMAs before staging z (overlap).
  if (wave == 0) {
    tdm_load_chunk(bboff0, wbf);
    if (nchunks > 1) tdm_load_chunk(bboff1, wbf + (size_t)16 * D);
  }

  // Stage + convert the z tile: 128 rows * 64 float4 over 128 threads.
  for (int i = t; i < 128 * (D / 4); i += 128) {
    const int r  = i >> 6;                    // 64 float4 per row
    const int c4 = (i & 63) * 4;
    const float4 f = ((const float4*)(z + (size_t)(row0 + r) * D))[i & 63];
    unsigned* dst = (unsigned*)&zt[r * SROW + c4];
    dst[0] = (unsigned)f2bf(f.x) | ((unsigned)f2bf(f.y) << 16);
    dst[1] = (unsigned)f2bf(f.z) | ((unsigned)f2bf(f.w) << 16);
  }
  __syncthreads();

  const int rbase = wave * 32;
  // A-fragment layout per ISA: lane(m=nl,h=half) elems 0-7 -> K=kk+8h..,
  // elems 8-15 -> K=kk+16+8h..  Two tiles: rows rbase+m and rbase+16+m.
  Frag a0[8], a1[8];
#pragma unroll
  for (int k = 0; k < 8; ++k) {
    const int co = k * 32 + 8 * half;
    a0[k].q[0] = *(const Q4*)&zt[(rbase + nl) * SROW + co];
    a0[k].q[1] = *(const Q4*)&zt[(rbase + nl) * SROW + co + 16];
    a1[k].q[0] = *(const Q4*)&zt[(rbase + 16 + nl) * SROW + co];
    a1[k].q[1] = *(const Q4*)&zt[(rbase + 16 + nl) * SROW + co + 16];
  }

  float best0[8], best1[8]; int bidx0[8], bidx1[8];
#pragma unroll
  for (int r = 0; r < 8; ++r) {
    best0[r] = 3.4e38f; best1[r] = 3.4e38f; bidx0[r] = 0; bidx1[r] = 0;
  }

  for (int c = 0; c < nchunks; ++c) {
    // Wave 0 confirms chunk c's DMA retired (TDM is in-order per wave),
    // then the barrier publishes the buffer to all waves.
    if (wave == 0) {
      if (c + 1 < nchunks) __builtin_amdgcn_s_wait_tensorcnt(1);
      else                 __builtin_amdgcn_s_wait_tensorcnt(0);
    }
    __syncthreads();

    const unsigned short* bbuf = (c & 1) ? bb1 : bb0;
    const int code = c * 16 + nl;             // this lane's codebook column
    const float wn = wnorm2[code];

    v8f acc0 = {}, acc1 = {};
#pragma unroll
    for (int k = 0; k < 8; ++k) {
      // B lane (n=nl,h=half): elems 0-7 -> K = kk + 8h + i, elems 8-15 -> K = kk + 16 + 8h + i
      Frag b;
      const Q4* bp = (const Q4*)(bbuf + nl * BROW + k * 32 + 8 * half);
      b.q[0] = bp[0];
      b.q[1] = bp[2];
      acc0 = __builtin_amdgcn_wmma_f32_16x16x32_bf16(
          false, a0[k].v, false, b.v, (short)0, acc0, false, false);
      acc1 = __builtin_amdgcn_wmma_f32_16x16x32_bf16(
          false, a1[k].v, false, b.v, (short)0, acc1, false, false);
    }

    // dist_n(row) = ||e_n||^2 - 2*dot  (||z||^2 constant per row -> dropped)
#pragma unroll
    for (int r = 0; r < 8; ++r) {
      const float d0 = wn - 2.0f * acc0[r];
      if (d0 < best0[r]) { best0[r] = d0; bidx0[r] = code; }
      const float d1 = wn - 2.0f * acc1[r];
      if (d1 < best1[r]) { best1[r] = d1; bidx1[r] = code; }
    }

    __syncthreads();   // all waves done reading this buffer
    if (wave == 0 && c + 2 < nchunks)
      tdm_load_chunk((c & 1) ? bboff1 : bboff0,
                     wbf + (size_t)(c + 2) * 16 * D);
  }

  // Reduce across the 16 lanes holding the same row (first-min tie-break).
#pragma unroll
  for (int r = 0; r < 8; ++r) {
    float d = best0[r]; int bi = bidx0[r];
    float e = best1[r]; int ei = bidx1[r];
#pragma unroll
    for (int off = 1; off < 16; off <<= 1) {
      const float od = __shfl_xor(d, off, 32);
      const int   oi = __shfl_xor(bi, off, 32);
      if (od < d || (od == d && oi < bi)) { d = od; bi = oi; }
      const float oe = __shfl_xor(e, off, 32);
      const int   o2 = __shfl_xor(ei, off, 32);
      if (oe < e || (oe == e && o2 < ei)) { e = oe; ei = o2; }
    }
    if (nl == 0) {
      idxout[row0 + rbase + 8 * half + r]      = bi;
      idxout[row0 + rbase + 16 + 8 * half + r] = ei;
    }
  }
}

// ---------------- Kernel C: gather z_q, indices, per-row SE -------------
__global__ void vq_gather(const float* __restrict__ z,
                          const float* __restrict__ w,
                          const int* __restrict__ idx,
                          float* __restrict__ out,
                          float* __restrict__ partial, int N) {
  __shared__ float sm[8];
  const int n = blockIdx.x, t = threadIdx.x;
  const int k = idx[n];
  const float wv = w[(size_t)k * D + t];
  const float zv = z[(size_t)n * D + t];
  out[1 + (size_t)n * D + t] = wv;            // z_q_st forward value == z_q
  const float df = wv - zv;
  const float s = blockSum(df * df, sm);
  if (t == 0) {
    partial[n] = s;
    out[1 + (size_t)N * D + n] = (float)k;    // indices, flattened after z_q
  }
}

// ---------------- Kernel D: deterministic loss reduction ----------------
__global__ void vq_finalize(const float* __restrict__ partial,
                            float* __restrict__ out, int N) {
  __shared__ float sm[8];
  float s = 0.0f;
  for (int i = threadIdx.x; i < N; i += 256) s += partial[i];
  const float tot = blockSum(s, sm);
  if (threadIdx.x == 0) out[0] = VQ_BETA * tot / ((float)N * (float)D);
}

extern "C" void kernel_launch(void* const* d_in, const int* in_sizes, int n_in,
                              void* d_out, int out_size, void* d_ws, size_t ws_size,
                              hipStream_t stream) {
  const float* z = (const float*)d_in[0];
  const float* w = (const float*)d_in[1];
  const int N = in_sizes[0] / D;   // 65536
  const int K = in_sizes[1] / D;   // 1024
  float* out = (float*)d_out;

  char* wsb = (char*)d_ws;
  unsigned short* wbf = (unsigned short*)wsb;           // K*D bf16
  size_t off = (size_t)K * D * sizeof(unsigned short);
  float* wnorm2 = (float*)(wsb + off);  off += (size_t)K * sizeof(float);
  int* idx      = (int*)(wsb + off);    off += (size_t)N * sizeof(int);
  float* partial = (float*)(wsb + off);                 // N floats

  vq_prep_weights<<<K, 256, 0, stream>>>(w, wbf, wnorm2);
  vq_argmin<<<N / 128, 128, LDS_TOTAL_B, stream>>>(z, wbf, wnorm2, idx, K);
  vq_gather<<<N, 256, 0, stream>>>(z, w, idx, out, partial, N);
  vq_finalize<<<1, 256, 0, stream>>>(partial, out, N);
}
